// CasualSelfAttention_56255481643139
// MI455X (gfx1250) — matrix operand order repaired and checked by reading, not compiled, not run on hardware
//
#include <hip/hip_runtime.h>
#include <hip/hip_bf16.h>
#include <math.h>

#define B_  4
#define T_  2048
#define C_  1024
#define H_  16
#define HD_ 64
#define N3C 3072
#define M_  (B_*T_)   // 8192

typedef __attribute__((ext_vector_type(16))) __bf16 v16bf;
typedef __attribute__((ext_vector_type(8)))  __bf16 v8bf;
typedef __attribute__((ext_vector_type(8)))  float  v8f;
typedef __attribute__((ext_vector_type(4)))  unsigned int v4u;
typedef __attribute__((ext_vector_type(8)))  int v8i;
typedef __attribute__((ext_vector_type(4)))  int v4i;

union Frag { v16bf v; v8bf h[2]; };

__device__ __forceinline__ v8f zero8() {
  v8f z;
  #pragma unroll
  for (int i = 0; i < 8; ++i) z[i] = 0.0f;
  return z;
}

// A-operand 16x32 bf16 (row-major [m][k], leading dim ld).
__device__ __forceinline__ v16bf load_a_frag(const __bf16* base, int ld) {
  const int lane = threadIdx.x & 31;
  const int m  = lane & 15;
  const int kh = (lane >> 4) * 8;
  const __bf16* p = base + (size_t)m * ld;
  Frag f;
  f.h[0] = *(const v8bf*)(p + kh);
  f.h[1] = *(const v8bf*)(p + 16 + kh);
  return f.v;
}

// B-operand 32x16 bf16 from N-major storage Bt[n][k] (leading dim ld).
__device__ __forceinline__ v16bf load_b_frag(const __bf16* base, int ld) {
  const int lane = threadIdx.x & 31;
  const int n  = lane & 15;
  const int kb = (lane >> 4) * 8;
  const __bf16* p = base + (size_t)n * ld + kb;
  Frag f;
  f.h[0] = *(const v8bf*)(p);
  f.h[1] = *(const v8bf*)(p + 16);
  return f.v;
}

__device__ __forceinline__ v8f wmma_bf16(v16bf a, v16bf b, v8f c) {
  return __builtin_amdgcn_wmma_f32_16x16x32_bf16(false, a, false, b, (short)0, c, false, false);
}

// ---- TDM: DMA a 128-row x 32-col bf16 tile (row stride Kdim elems) into LDS.
// LDS rows are padded 64B data + 16B pad => 80B rows ([128][40] bf16 layout):
//   pad_interval code 3 -> pad after every 8 x 8B = 64B; pad_amount code 3 -> 4 DWORDs = 16B.
__device__ __forceinline__ void tdm_load_tile(const __bf16* src, unsigned lds_off,
                                              int Kdim, int dim1) {
  unsigned long long ga = (unsigned long long)(size_t)src;
  v4u g0;
  g0[0] = 1u;                                        // count=1, user descriptor
  g0[1] = lds_off;                                   // lds_addr
  g0[2] = (unsigned)(ga & 0xFFFFFFFFu);              // global_addr[31:0]
  g0[3] = (unsigned)((ga >> 32) & 0x1FFFFFFu)        // global_addr[56:32]
          | 0x80000000u;                             // type=2 ("image")
  v8i g1;
  g1[0] = (int)((1u << 16)                           // data_size = 2 bytes
              | (1u << 20)                           // pad_enable
              | (3u << 22)                           // pad_interval: 64B
              | (3u << 25));                         // pad_amount : 16B
  g1[1] = (int)((unsigned)(Kdim & 0xFFFF) << 16);    // tensor_dim0[15:0]
  g1[2] = (int)(((unsigned)Kdim >> 16)               // tensor_dim0[31:16]
              | ((unsigned)(dim1 & 0xFFFF) << 16));  // tensor_dim1[15:0]
  g1[3] = (int)(((unsigned)dim1 >> 16)               // tensor_dim1[31:16]
              | (32u << 16));                        // tile_dim0 = 32
  g1[4] = 128;                                       // tile_dim1 = 128, tile_dim2 = 0
  g1[5] = Kdim;                                      // tensor_dim0_stride[31:0]
  g1[6] = 0;
  g1[7] = 0;
  v4i gz = {0, 0, 0, 0};
#if __clang_major__ >= 23
  v8i gz8 = {0, 0, 0, 0, 0, 0, 0, 0};
  __builtin_amdgcn_tensor_load_to_lds(g0, g1, gz, gz, gz8, 0);
#else
  __builtin_amdgcn_tensor_load_to_lds(g0, g1, gz, gz, 0);
#endif
}

// ---------------- conversion kernels ----------------
__global__ void cvt_bf16_kernel(const float* __restrict__ src, __bf16* __restrict__ dst, int n) {
  int i = blockIdx.x * blockDim.x + threadIdx.x;
  if (i < n) dst[i] = (__bf16)src[i];
}

// src [rows][cols] fp32  ->  dst [cols][rows] bf16 (transposed)
__global__ void cvt_transpose_kernel(const float* __restrict__ src, __bf16* __restrict__ dst,
                                     int rows, int cols) {
  int i = blockIdx.x * blockDim.x + threadIdx.x;
  if (i < rows * cols) {
    int n = i / rows;
    int k = i % rows;
    dst[(size_t)n * rows + k] = (__bf16)src[(size_t)k * cols + n];
  }
}

// ---------------- tiled WMMA GEMM with TDM double-buffered LDS staging ----------------
// C = A(bf16 [M][K]) * Bt(bf16 [N][K])^T + bias
// Dynamic LDS: 2 buffers x (A-tile 10240B + B-tile 10240B) = 40960 bytes.
// MODE 0: bf16 out + V slab transposed into Vt[b,h,d,t]; MODE 1: fp32 out.
template <int MODE>
__global__ __launch_bounds__(256) void gemm_kernel(
    const __bf16* __restrict__ A, const __bf16* __restrict__ Bt,
    const float* __restrict__ bias,
    __bf16* __restrict__ outBf, __bf16* __restrict__ Vt, float* __restrict__ outF,
    int Mdim, int Ndim, int Kdim)
{
  extern __shared__ char smem[];

  const int nTilesN = Ndim / 128;
  const int m0 = (blockIdx.x / nTilesN) * 128;
  const int n0 = (blockIdx.x % nTilesN) * 128;
  const int tid  = threadIdx.x;
  const int wave = tid >> 5;
  const int wm = wave >> 1;          // 0..3  (32 rows each)
  const int wn = wave & 1;           // 0..1  (64 cols each)

  v8f acc[2][4];
  #pragma unroll
  for (int i = 0; i < 2; ++i)
    #pragma unroll
    for (int j = 0; j < 4; ++j) acc[i][j] = zero8();

  const int nIter = Kdim / 32;
  if (wave == 0) {   // prologue: fill buffer 0
    tdm_load_tile(A  + (size_t)m0 * Kdim, 0u,     Kdim, Mdim);
    tdm_load_tile(Bt + (size_t)n0 * Kdim, 10240u, Kdim, Ndim);
  }

  for (int it = 0; it < nIter; ++it) {
    __syncthreads();   // everyone done reading buffer (it+1)&1 from iter it-1
    if (wave == 0) {
      if (it + 1 < nIter) {
        const unsigned nb = (unsigned)((it + 1) & 1) * 20480u;
        tdm_load_tile(A  + (size_t)m0 * Kdim + (it + 1) * 32, nb,          Kdim, Mdim);
        tdm_load_tile(Bt + (size_t)n0 * Kdim + (it + 1) * 32, nb + 10240u, Kdim, Ndim);
        __builtin_amdgcn_s_wait_tensorcnt((short)2);  // current tile's 2 DMAs done
      } else {
        __builtin_amdgcn_s_wait_tensorcnt((short)0);
      }
    }
    __syncthreads();   // current buffer visible to all waves

    const __bf16* As = (const __bf16*)(smem + (size_t)(it & 1) * 20480u);
    const __bf16* Bs = As + 5120;    // +10240 bytes

    v16bf af[2], bfr[4];
    af[0] = load_a_frag(As + (wm * 32)      * 40, 40);
    af[1] = load_a_frag(As + (wm * 32 + 16) * 40, 40);
    #pragma unroll
    for (int j = 0; j < 4; ++j)
      bfr[j] = load_b_frag(Bs + (wn * 64 + j * 16) * 40, 40);

    #pragma unroll
    for (int i = 0; i < 2; ++i)
      #pragma unroll
      for (int j = 0; j < 4; ++j)
        acc[i][j] = wmma_bf16(af[i], bfr[j], acc[i][j]);
  }

  const int lane = tid & 31;
  const int colL = lane & 15;
  const int rowH = (lane >> 4) * 8;
  #pragma unroll
  for (int i = 0; i < 2; ++i) {
    #pragma unroll
    for (int j = 0; j < 4; ++j) {
      const int nn = n0 + wn * 64 + j * 16 + colL;
      const float bv = bias[nn];
      #pragma unroll
      for (int r = 0; r < 8; ++r) {
        const int mm = m0 + wm * 32 + i * 16 + rowH + r;
        const float val = acc[i][j][r] + bv;
        if (MODE == 0) {
          outBf[(size_t)mm * N3C + nn] = (__bf16)val;
          if (nn >= 2 * C_) {                 // V slab -> transposed copy
            const int dall = nn - 2 * C_;
            const int hh = dall >> 6, dd = dall & 63;
            const int bb = mm >> 11, tt = mm & (T_ - 1);
            Vt[(((size_t)bb * H_ + hh) * HD_ + dd) * T_ + tt] = (__bf16)val;
          }
        } else {
          outF[(size_t)mm * C_ + nn] = val;
        }
      }
    }
  }
}

// ---------------- flash attention: 1 wave per (b, h, 16-query tile) ----------------
__global__ __launch_bounds__(128) void attn_kernel(
    const __bf16* __restrict__ qkv, const __bf16* __restrict__ Vt,
    __bf16* __restrict__ Y)
{
  __shared__ __bf16 Plds[4][16][32];   // per-wave P staging (acc layout -> A layout)
  const int tid  = threadIdx.x;
  const int lane = tid & 31;
  const int wv   = tid >> 5;
  const int gw   = blockIdx.x * 4 + wv;
  const int nQT  = T_ / 16;            // 128
  const int qt = gw % nQT;
  const int h  = (gw / nQT) % H_;
  const int b  = gw / (nQT * H_);
  const int q0 = qt * 16;

  const int colL   = lane & 15;
  const int halfId = lane >> 4;

  const __bf16* Qbase = qkv + (size_t)(b * T_ + q0) * N3C + h * HD_;
  const __bf16* Kbase = qkv + (size_t)b * T_ * N3C + C_ + h * HD_;
  const __bf16* Vbase = Vt + ((size_t)b * H_ + h) * HD_ * T_;

  v16bf aq[2];
  aq[0] = load_a_frag(Qbase, N3C);        // d = 0..31
  aq[1] = load_a_frag(Qbase + 32, N3C);   // d = 32..63

  float mrow[8], lrow[8];
  v8f o[4];
  #pragma unroll
  for (int j = 0; j < 4; ++j) o[j] = zero8();
  #pragma unroll
  for (int r = 0; r < 8; ++r) { mrow[r] = -1e30f; lrow[r] = 0.0f; }

  const int ktEnd = q0 + 16;
  for (int kt = 0; kt < ktEnd; kt += 32) {
    v8f s0 = zero8(), s1 = zero8();      // keys kt..+15 / kt+16..+31
    #pragma unroll
    for (int ks = 0; ks < 2; ++ks) {
      v16bf kb0 = load_b_frag(Kbase + (size_t)kt * N3C + ks * 32, N3C);
      v16bf kb1 = load_b_frag(Kbase + (size_t)(kt + 16) * N3C + ks * 32, N3C);
      s0 = wmma_bf16(aq[ks], kb0, s0);
      s1 = wmma_bf16(aq[ks], kb1, s1);
    }

    const bool fullTile = (kt + 31 <= q0);   // wave-uniform: no masking needed
    const int key0 = kt + colL;
    const int key1 = kt + 16 + colL;
    #pragma unroll
    for (int r = 0; r < 8; ++r) {
      float x0, x1;
      if (fullTile) {
        x0 = s0[r] * 0.125f;
        x1 = s1[r] * 0.125f;
      } else {
        const int qg = q0 + halfId * 8 + r;
        x0 = (key0 <= qg) ? s0[r] * 0.125f : -1e30f;
        x1 = (key1 <= qg) ? s1[r] * 0.125f : -1e30f;
      }
      float mx = fmaxf(x0, x1);
      #pragma unroll
      for (int off = 8; off >= 1; off >>= 1) mx = fmaxf(mx, __shfl_xor(mx, off, 16));
      const float mnew = fmaxf(mrow[r], mx);
      const float p0 = __expf(x0 - mnew);
      const float p1 = __expf(x1 - mnew);
      const float corr = __expf(mrow[r] - mnew);
      float rs = p0 + p1;
      #pragma unroll
      for (int off = 8; off >= 1; off >>= 1) rs += __shfl_xor(rs, off, 16);
      lrow[r] = lrow[r] * corr + rs;
      mrow[r] = mnew;
      #pragma unroll
      for (int j = 0; j < 4; ++j) o[j][r] *= corr;
      const int row = halfId * 8 + r;
      Plds[wv][row][colL]      = (__bf16)p0;
      Plds[wv][row][16 + colL] = (__bf16)p1;
    }
    asm volatile("s_wait_dscnt 0" ::: "memory");
    v16bf pf = load_a_frag(&Plds[wv][0][0], 32);
    #pragma unroll
    for (int j = 0; j < 4; ++j) {
      v16bf vf = load_b_frag(Vbase + (size_t)(j * 16) * T_ + kt, T_);
      o[j] = wmma_bf16(pf, vf, o[j]);
    }
  }

  #pragma unroll
  for (int r = 0; r < 8; ++r) {
    const float inv = 1.0f / lrow[r];
    const int row = b * T_ + q0 + halfId * 8 + r;
    #pragma unroll
    for (int j = 0; j < 4; ++j)
      Y[(size_t)row * C_ + h * HD_ + j * 16 + colL] = (__bf16)(o[j][r] * inv);
  }
}

// ---------------- host launcher ----------------
extern "C" void kernel_launch(void* const* d_in, const int* in_sizes, int n_in,
                              void* d_out, int out_size, void* d_ws, size_t ws_size,
                              hipStream_t stream) {
  const float* x      = (const float*)d_in[0];
  const float* W_attn = (const float*)d_in[1];
  const float* b_attn = (const float*)d_in[2];
  const float* W_proj = (const float*)d_in[3];
  const float* b_proj = (const float*)d_in[4];
  float* out = (float*)d_out;

  char* ws = (char*)d_ws;
  size_t off = 0;
  __bf16* x_bf    = (__bf16*)(ws + off); off += (size_t)M_ * C_ * 2;            // 16 MB
  __bf16* Wt_attn = (__bf16*)(ws + off); off += (size_t)N3C * C_ * 2;           //  6 MB
  __bf16* Wt_proj = (__bf16*)(ws + off); off += (size_t)C_ * C_ * 2;            //  2 MB
  __bf16* qkv     = (__bf16*)(ws + off); off += (size_t)M_ * N3C * 2;           // 48 MB
  __bf16* Vt      = (__bf16*)(ws + off); off += (size_t)B_ * H_ * HD_ * T_ * 2; // 16 MB
  __bf16* Ybf     = (__bf16*)(ws + off); off += (size_t)M_ * C_ * 2;            // 16 MB

  cvt_bf16_kernel<<<(M_ * C_) / 256, 256, 0, stream>>>(x, x_bf, M_ * C_);
  cvt_transpose_kernel<<<(C_ * N3C) / 256, 256, 0, stream>>>(W_attn, Wt_attn, C_, N3C);
  cvt_transpose_kernel<<<(C_ * C_) / 256, 256, 0, stream>>>(W_proj, Wt_proj, C_, C_);

  gemm_kernel<0><<<(M_ / 128) * (N3C / 128), 256, 40960, stream>>>(
      x_bf, Wt_attn, b_attn, qkv, Vt, nullptr, M_, N3C, C_);

  attn_kernel<<<(B_ * H_ * (T_ / 16)) / 4, 128, 0, stream>>>(qkv, Vt, Ybf);

  gemm_kernel<1><<<(M_ / 128) * (C_ / 128), 256, 40960, stream>>>(
      Ybf, Wt_proj, b_proj, nullptr, nullptr, out, M_, C_, C_);
}
